// SmallTemporalAttention_18339510354200
// MI455X (gfx1250) — matrix operand order repaired and checked by reading, not compiled, not run on hardware
//
#include <hip/hip_runtime.h>

#define B_ 8
#define C_ 64
#define T_ 128
#define N_ 512
#define H_ 32
#define NSEQ 4
#define NTHREADS 256

typedef _Float16 half16 __attribute__((ext_vector_type(16)));
typedef float    v8f    __attribute__((ext_vector_type(8)));

union HF {
  half16 v;
  unsigned u[8];
  uint4 q[2];
};

static __device__ __forceinline__ v8f vzero() {
  v8f z = {0.f, 0.f, 0.f, 0.f, 0.f, 0.f, 0.f, 0.f};
  return z;
}

static __device__ __forceinline__ unsigned pkh2(float a, float b) {
  union { _Float16 h[2]; unsigned u; } x;
  x.h[0] = (_Float16)a;
  x.h[1] = (_Float16)b;
  return x.u;
}

static __device__ __forceinline__ v8f wmma_f16(half16 a, half16 b, v8f c) {
  // D = A(16x32 f16) * B(32x16 f16) + C(16x16 f32)
  return __builtin_amdgcn_wmma_f32_16x16x32_f16(false, a, false, b, (short)0, c, false, false);
}

// A fragment (16x32, f16) assembled from an f32 LDS row (row-major, 32B-aligned regions).
// lane: M = lane&15 ; h2 = lane>>4 ; VGPR v<4: K=kBase+2v+8*h2 ; v>=4: K=kBase+16+2(v-4)+8*h2
static __device__ __forceinline__ half16 load_a_f32(const float* row, int kBase, int h2) {
  const float4* p0 = (const float4*)(row + kBase + 8 * h2);
  const float4* p1 = (const float4*)(row + kBase + 16 + 8 * h2);
  float4 a = p0[0], b = p0[1], c = p1[0], d = p1[1];
  HF r;
  r.u[0] = pkh2(a.x, a.y); r.u[1] = pkh2(a.z, a.w);
  r.u[2] = pkh2(b.x, b.y); r.u[3] = pkh2(b.z, b.w);
  r.u[4] = pkh2(c.x, c.y); r.u[5] = pkh2(c.z, c.w);
  r.u[6] = pkh2(d.x, d.y); r.u[7] = pkh2(d.z, d.w);
  return r.v;
}

// A fragment (16x32, f16) from a per-wave [16][32] f16 staging buffer.
static __device__ __forceinline__ half16 load_a_f16(const _Float16* base, int qn, int h2) {
  const _Float16* row = base + qn * 32;
  HF r;
  r.q[0] = *(const uint4*)(row + 8 * h2);
  r.q[1] = *(const uint4*)(row + 16 + 8 * h2);
  return r.v;
}

// B fragment (32x16, f16). base layout: [n][k], row stride rs (elements).
// lane: N = nBase + (lane&15) ; VGPR v<4: K = kBase + 8*(lane>>4) + 2v (+1) ; v>=4: K = kBase + 16 + 8*(lane>>4) + 2(v-4) (+1)
static __device__ __forceinline__ half16 load_b_f16(const _Float16* base, int rs, int nBase,
                                                    int kBase, int qn, int h2) {
  const _Float16* p = base + (nBase + qn) * rs + kBase + 8 * h2;
  HF r;
  r.q[0] = *(const uint4*)p;
  r.q[1] = *(const uint4*)(p + 16);
  return r.v;
}

// B fragment (32x16) built straight from a global f32 weight matrix stored [K][N] row-major
// with leading dimension ld. One-time cost per wave; weights are tiny and L2/L0 resident.
static __device__ __forceinline__ half16 load_b_f32g(const float* w, int ld, int nBase,
                                                     int kBase, int qn, int h2) {
  const float* p = w + (long)(kBase + 8 * h2) * ld + nBase + qn;
  HF r;
  #pragma unroll
  for (int v = 0; v < 8; ++v)
    r.u[v] = pkh2(p[(2 * v + (v >= 4 ? 8 : 0)) * ld], p[(2 * v + 1 + (v >= 4 ? 8 : 0)) * ld]);
  return r.v;
}

__global__ __launch_bounds__(NTHREADS) void sta_kernel(
    const float* __restrict__ x,
    const float* __restrict__ wq, const float* __restrict__ bq,
    const float* __restrict__ wk, const float* __restrict__ bk,
    const float* __restrict__ wv, const float* __restrict__ bv,
    const float* __restrict__ wg, const float* __restrict__ bg,
    const float* __restrict__ wo, const float* __restrict__ bo,
    const float* __restrict__ gamma, const float* __restrict__ beta,
    float* __restrict__ out)
{
  __shared__ float    Xf[NSEQ][T_][C_];   // 128 KB: x tiles (f32), later reused as Y
  __shared__ _Float16 Ksh[T_][H_];        // 8 KB : K row-major (B operand of Q@K^T)
  __shared__ _Float16 Vt[H_][T_];         // 8 KB : V transposed (B operand of P@V)
  __shared__ _Float16 Stage[8][16][32];   // 8 KB : per-wave A-operand staging (Q/P/O/gated)
  __shared__ float bqs[H_], bks[H_], bvs[H_], bgs[H_];
  __shared__ float bos[C_], gms[C_], bts[C_];
  // total ~153.3 KB -> one block per WGP, ~512 VGPRs/wave budget

  const int tid  = threadIdx.x;
  const int wid  = tid >> 5;
  const int lane = tid & 31;
  const int h2   = lane >> 4;
  const int qn   = lane & 15;

  const int b  = blockIdx.x >> 7;           // N_/NSEQ = 128 blocks per b
  const int n0 = (blockIdx.x & 127) * NSEQ;

  if (tid < H_) { bqs[tid] = bq[tid]; bks[tid] = bk[tid]; bvs[tid] = bv[tid]; bgs[tid] = bg[tid]; }
  if (tid < C_) { bos[tid] = bo[tid]; gms[tid] = gamma[tid]; bts[tid] = beta[tid]; }

  // ---- gather x: one float4 covers NSEQ=4 consecutive n (quarters L2 transaction count) ----
  for (int i = tid; i < T_ * C_; i += NTHREADS) {
    int t = i >> 6, c = i & 63;
    const float4 v4 = *(const float4*)(x + (((long)b * C_ + c) * T_ + t) * N_ + n0);
    Xf[0][t][c] = v4.x;
    Xf[1][t][c] = v4.y;
    Xf[2][t][c] = v4.z;
    Xf[3][t][c] = v4.w;
  }

  // ---- weight B-fragments straight from global into registers (sequence-invariant) ----
  half16 wqf[2][2], wkf[2][2], wvf[2][2], wgf[2], wof[4];
  #pragma unroll
  for (int kc = 0; kc < 2; ++kc) {
    #pragma unroll
    for (int tn = 0; tn < 2; ++tn) {
      wqf[kc][tn] = load_b_f32g(wq, H_, 16 * tn, 32 * kc, qn, h2);
      wkf[kc][tn] = load_b_f32g(wk, H_, 16 * tn, 32 * kc, qn, h2);
      wvf[kc][tn] = load_b_f32g(wv, H_, 16 * tn, 32 * kc, qn, h2);
    }
  }
  wgf[0] = load_b_f32g(wg, H_,  0, 0, qn, h2);
  wgf[1] = load_b_f32g(wg, H_, 16, 0, qn, h2);
  #pragma unroll
  for (int tn = 0; tn < 4; ++tn)
    wof[tn] = load_b_f32g(wo, C_, 16 * tn, 0, qn, h2);

  __syncthreads();

  const int rowB = 16 * wid;                 // wave owns rows [rowB, rowB+16)
  const float qscale = 0.17677669529663687f; // 1/sqrt(H)

  for (int s = 0; s < NSEQ; ++s) {
    // ======== QKV projections (A from f32 X, B weights already in registers) ========
    v8f qa[2], ka[2], va[2];
    #pragma unroll
    for (int tn = 0; tn < 2; ++tn) { qa[tn] = vzero(); ka[tn] = vzero(); va[tn] = vzero(); }
    half16 af[2];
    #pragma unroll
    for (int kc = 0; kc < 2; ++kc)
      af[kc] = load_a_f32(&Xf[s][rowB + qn][0], 32 * kc, h2);
    #pragma unroll
    for (int kc = 0; kc < 2; ++kc) {
      #pragma unroll
      for (int tn = 0; tn < 2; ++tn) {
        qa[tn] = wmma_f16(af[kc], wqf[kc][tn], qa[tn]);
        ka[tn] = wmma_f16(af[kc], wkf[kc][tn], ka[tn]);
        va[tn] = wmma_f16(af[kc], wvf[kc][tn], va[tn]);
      }
    }
    // Q (pre-scaled) -> own Stage ; K -> Ksh (row major) ; V -> Vt (transposed)
    #pragma unroll
    for (int tn = 0; tn < 2; ++tn) {
      int col = 16 * tn + qn;
      float bqv = bqs[col], bkv = bks[col], bvv = bvs[col];
      #pragma unroll
      for (int j = 0; j < 8; ++j) {
        Stage[wid][j + 8 * h2][col] = (_Float16)((qa[tn][j] + bqv) * qscale);
        Ksh[rowB + j + 8 * h2][col] = (_Float16)(ka[tn][j] + bkv);
      }
      #pragma unroll
      for (int j = 0; j < 8; j += 2) {
        unsigned pv = pkh2(va[tn][j] + bvv, va[tn][j + 1] + bvv);
        *(unsigned*)&Vt[col][rowB + 8 * h2 + j] = pv;
      }
    }
    half16 qfrag = load_a_f16(&Stage[wid][0][0], qn, h2);  // wave-local, in-order DS
    __syncthreads();  // K, V visible to all waves

    // ======== S = Q K^T: preload all 8 K fragments, then straight WMMA chain ========
    half16 kf[8];
    #pragma unroll
    for (int tn = 0; tn < 8; ++tn)
      kf[tn] = load_b_f16(&Ksh[0][0], H_, 16 * tn, 0, qn, h2);
    v8f sc[8];
    #pragma unroll
    for (int tn = 0; tn < 8; ++tn)
      sc[tn] = wmma_f16(qfrag, kf[tn], vzero());

    // position bias + softmax (rows are wave-local; width-16 shfl reductions)
    float mx[8], sm[8];
    #pragma unroll
    for (int j = 0; j < 8; ++j) mx[j] = -1e30f;
    #pragma unroll
    for (int tn = 0; tn < 8; ++tn) {
      #pragma unroll
      for (int j = 0; j < 8; ++j) {
        int row = rowB + j + 8 * h2;
        int col = 16 * tn + qn;
        float v = sc[tn][j] + __expf(-fabsf((float)(row - col)) * (1.0f / 128.0f));
        sc[tn][j] = v;
        mx[j] = fmaxf(mx[j], v);
      }
    }
    #pragma unroll
    for (int j = 0; j < 8; ++j) {
      mx[j] = fmaxf(mx[j], __shfl_xor(mx[j], 1, 32));
      mx[j] = fmaxf(mx[j], __shfl_xor(mx[j], 2, 32));
      mx[j] = fmaxf(mx[j], __shfl_xor(mx[j], 4, 32));
      mx[j] = fmaxf(mx[j], __shfl_xor(mx[j], 8, 32));
      sm[j] = 0.f;
    }
    #pragma unroll
    for (int tn = 0; tn < 8; ++tn) {
      #pragma unroll
      for (int j = 0; j < 8; ++j) {
        float e = __expf(sc[tn][j] - mx[j]);
        sc[tn][j] = e;
        sm[j] += e;
      }
    }
    #pragma unroll
    for (int j = 0; j < 8; ++j) {
      sm[j] += __shfl_xor(sm[j], 1, 32);
      sm[j] += __shfl_xor(sm[j], 2, 32);
      sm[j] += __shfl_xor(sm[j], 4, 32);
      sm[j] += __shfl_xor(sm[j], 8, 32);
      sm[j] = 1.0f / sm[j];
    }

    // ======== O = P V: preload all 8 V fragments, then chunk P through staging ========
    half16 vf[2][4];
    #pragma unroll
    for (int kc = 0; kc < 4; ++kc) {
      vf[0][kc] = load_b_f16(&Vt[0][0], T_,  0, 32 * kc, qn, h2);
      vf[1][kc] = load_b_f16(&Vt[0][0], T_, 16, 32 * kc, qn, h2);
    }
    v8f oa[2];
    oa[0] = vzero(); oa[1] = vzero();
    #pragma unroll
    for (int kc = 0; kc < 4; ++kc) {
      #pragma unroll
      for (int tt = 0; tt < 2; ++tt) {
        int tn = 2 * kc + tt;
        #pragma unroll
        for (int j = 0; j < 8; ++j)
          Stage[wid][j + 8 * h2][16 * tt + qn] = (_Float16)(sc[tn][j] * sm[j]);
      }
      half16 pf = load_a_f16(&Stage[wid][0][0], qn, h2);
      oa[0] = wmma_f16(pf, vf[0][kc], oa[0]);
      oa[1] = wmma_f16(pf, vf[1][kc], oa[1]);
    }

    // ======== gate: out *= sigmoid(out @ Wg + bg) ========
    #pragma unroll
    for (int tn = 0; tn < 2; ++tn)
      #pragma unroll
      for (int j = 0; j < 8; ++j)
        Stage[wid][j + 8 * h2][16 * tn + qn] = (_Float16)oa[tn][j];
    half16 of = load_a_f16(&Stage[wid][0][0], qn, h2);
    v8f ga[2];
    ga[0] = wmma_f16(of, wgf[0], vzero());
    ga[1] = wmma_f16(of, wgf[1], vzero());
    #pragma unroll
    for (int tn = 0; tn < 2; ++tn) {
      float bgv = bgs[16 * tn + qn];
      #pragma unroll
      for (int j = 0; j < 8; ++j) {
        float g = ga[tn][j] + bgv;
        float gate = 1.0f / (1.0f + __expf(-g));
        Stage[wid][j + 8 * h2][16 * tn + qn] = (_Float16)(oa[tn][j] * gate);
      }
    }
    half16 gf = load_a_f16(&Stage[wid][0][0], qn, h2);

    // ======== Y = gated @ Wo + bo + residual, LayerNorm over C ========
    v8f ya[4];
    #pragma unroll
    for (int tn = 0; tn < 4; ++tn)
      ya[tn] = wmma_f16(gf, wof[tn], vzero());

    float yv[4][8];
    float mu[8];
    #pragma unroll
    for (int j = 0; j < 8; ++j) mu[j] = 0.f;
    #pragma unroll
    for (int tn = 0; tn < 4; ++tn) {
      int col = 16 * tn + qn;
      float bov = bos[col];
      #pragma unroll
      for (int j = 0; j < 8; ++j) {
        int row = rowB + j + 8 * h2;
        float v = ya[tn][j] + bov + Xf[s][row][col];  // exact f32 residual
        yv[tn][j] = v;
        mu[j] += v;
      }
    }
    #pragma unroll
    for (int j = 0; j < 8; ++j) {
      mu[j] += __shfl_xor(mu[j], 1, 32);
      mu[j] += __shfl_xor(mu[j], 2, 32);
      mu[j] += __shfl_xor(mu[j], 4, 32);
      mu[j] += __shfl_xor(mu[j], 8, 32);
      mu[j] *= (1.0f / 64.0f);
    }
    float var[8];
    #pragma unroll
    for (int j = 0; j < 8; ++j) var[j] = 0.f;
    #pragma unroll
    for (int tn = 0; tn < 4; ++tn)
      #pragma unroll
      for (int j = 0; j < 8; ++j) {
        float d = yv[tn][j] - mu[j];
        var[j] += d * d;
      }
    #pragma unroll
    for (int j = 0; j < 8; ++j) {
      var[j] += __shfl_xor(var[j], 1, 32);
      var[j] += __shfl_xor(var[j], 2, 32);
      var[j] += __shfl_xor(var[j], 4, 32);
      var[j] += __shfl_xor(var[j], 8, 32);
      var[j] = rsqrtf(var[j] * (1.0f / 64.0f) + 1e-5f);
    }
    #pragma unroll
    for (int tn = 0; tn < 4; ++tn) {
      int col = 16 * tn + qn;
      float gmv = gms[col], btv = bts[col];
      #pragma unroll
      for (int j = 0; j < 8; ++j) {
        int row = rowB + j + 8 * h2;
        Xf[s][row][col] = (yv[tn][j] - mu[j]) * var[j] * gmv + btv;  // Y in place
      }
    }
    __syncthreads();  // protect Ksh/Vt reuse next seq + Xf completeness
  }

  // ---- scatter output: one float4 covers the NSEQ=4 n's (16B stores) ----
  for (int i = tid; i < T_ * C_; i += NTHREADS) {
    int t = i >> 6, c = i & 63;
    float4 v4;
    v4.x = Xf[0][t][c];
    v4.y = Xf[1][t][c];
    v4.z = Xf[2][t][c];
    v4.w = Xf[3][t][c];
    *(float4*)(out + (((long)b * C_ + c) * T_ + t) * N_ + n0) = v4;
  }
}

extern "C" void kernel_launch(void* const* d_in, const int* in_sizes, int n_in,
                              void* d_out, int out_size, void* d_ws, size_t ws_size,
                              hipStream_t stream) {
  (void)in_sizes; (void)n_in; (void)out_size; (void)d_ws; (void)ws_size;
  const float* x     = (const float*)d_in[0];
  const float* wq    = (const float*)d_in[1];
  const float* bq    = (const float*)d_in[2];
  const float* wk    = (const float*)d_in[3];
  const float* bk    = (const float*)d_in[4];
  const float* wv    = (const float*)d_in[5];
  const float* bv    = (const float*)d_in[6];
  const float* wg    = (const float*)d_in[7];
  const float* bg    = (const float*)d_in[8];
  const float* wo    = (const float*)d_in[9];
  const float* bo    = (const float*)d_in[10];
  const float* gamma = (const float*)d_in[11];
  const float* beta  = (const float*)d_in[12];
  float* out = (float*)d_out;

  dim3 grid(B_ * (N_ / NSEQ));  // 1024 blocks, one per (b, n-quad)
  sta_kernel<<<grid, NTHREADS, 0, stream>>>(x, wq, bq, wk, bk, wv, bv,
                                            wg, bg, wo, bo, gamma, beta, out);
}
